// TreeLSTMBranch_53506702573727
// MI455X (gfx1250) — matrix operand order repaired and checked by reading, not compiled, not run on hardware
//
#include <hip/hip_runtime.h>
#include <hip/hip_bf16.h>

// ---------------------------------------------------------------------------
// TreeLSTM double-sweep for MI455X (gfx1250, wave32, WMMA bf16 16x16x32).
//
//   - 64 blocks x 512 threads (16 waves); block = one 16-row batch chain.
//   - wave w owns hidden columns [w*16, w*16+16): one 16x16 accumulator tile
//     each for i / o / u / f, cell state c register-resident in D-layout.
//   - h (16x256 bf16) exchanged through LDS each step; weights stream from
//     L2 via per-lane base pointers + immediate offsets (no per-load math).
//   - fused linear score each forward-sweep step; atomic-binned mask-mean
//     + argmax epilogue.
// ---------------------------------------------------------------------------

typedef __bf16 bf16_t;
typedef __attribute__((ext_vector_type(16))) __bf16 v16bf;
typedef __attribute__((ext_vector_type(8)))  float  v8f;

#define B_  1024
#define L_  64
#define X_  128
#define H_  256
#define H3_ 768
#define K_  32

union BF16Frag { uint4 u[2]; v16bf v; };

__device__ __forceinline__ v8f wmma_bf16(v16bf a, v16bf b, v8f c) {
  return __builtin_amdgcn_wmma_f32_16x16x32_bf16(false, a, false, b, (short)0, c,
                                                 false, false);
}

// B fragment (32x16): lane holds one weight row (== one B column); the
// per-lane base pointer already includes row base + (lane>>4)*8 K-offset;
// halves at K+0..7 and K+16..23.
__device__ __forceinline__ v16bf load_b_at(const bf16_t* p, int kk) {
  const bf16_t* q = p + kk * 32;
  BF16Frag f;
  f.u[0] = *(const uint4*)(q);
  f.u[1] = *(const uint4*)(q + 16);
  return f.v;
}

// A fragment (16x32): lane holds row lane&15; base includes (lane>>4)*8;
// halves at K+0..7 and K+16..23.
__device__ __forceinline__ v16bf load_a_at(const bf16_t* p, int kk) {
  const bf16_t* q = p + kk * 32;
  BF16Frag f;
  f.u[0] = *(const uint4*)(q);
  f.u[1] = *(const uint4*)(q + 16);
  return f.v;
}

__device__ __forceinline__ float sigm(float x) {
  return 1.0f / (1.0f + __expf(-x));
}

// ---------------------------------------------------------------------------
__global__ void cvt_f32_bf16(const float* __restrict__ s, bf16_t* __restrict__ d,
                             int n) {
  for (int i = blockIdx.x * blockDim.x + threadIdx.x; i < n;
       i += gridDim.x * blockDim.x)
    d[i] = (bf16_t)s[i];
}

__global__ void zero_bins(float* bins) {
  int t = threadIdx.x;
  if (t < 192) bins[t] = 0.0f;
}

// ---------------------------------------------------------------------------
__global__ __launch_bounds__(512) void treelstm_sweep(
    const bf16_t* __restrict__ featB, const bf16_t* __restrict__ WiouB,
    const bf16_t* __restrict__ UiouB, const bf16_t* __restrict__ WfB,
    const bf16_t* __restrict__ UfB, const float* __restrict__ iou_in,
    const float* __restrict__ c_in, const float* __restrict__ b_iou,
    const float* __restrict__ bW_f, const float* __restrict__ b_f,
    const float* __restrict__ bU_f, const float* __restrict__ lin_w,
    const float* __restrict__ lin_b, float* __restrict__ hsc) {
  __shared__ bf16_t hL[16 * H_];
  __shared__ float  red[16][17];
  __shared__ float  lwL[H_];

  const int tid    = threadIdx.x;
  const int wid    = tid >> 5;                 // 0..15
  const int lane   = tid & 31;
  const int b0     = blockIdx.x * 16;
  const int n0     = wid * 16 + (lane & 15);   // this wave's hidden column
  const int rowOff = (lane >> 4) * 8;          // C/D: vgpr r -> row r+rowOff

  if (tid < H_) lwL[tid] = lin_w[tid];
  const float linb = lin_b[0];

  const float bi = b_iou[n0];
  const float bo = b_iou[H_ + n0];
  const float bu = b_iou[2 * H_ + n0];
  const float bf = bW_f[n0] + bU_f[n0] + b_f[n0];

  // ---- per-lane base pointers (hoisted address math) ----
  const int kb16 = (lane >> 4) * 8;            // B fragment K base
  const bf16_t* pUi = UiouB + (size_t)(n0)*H_ + kb16;
  const bf16_t* pUo = UiouB + (size_t)(H_ + n0) * H_ + kb16;
  const bf16_t* pUu = UiouB + (size_t)(2 * H_ + n0) * H_ + kb16;
  const bf16_t* pUf = UfB + (size_t)(n0)*H_ + kb16;
  const bf16_t* pWi = WiouB + (size_t)(n0)*X_ + kb16;
  const bf16_t* pWo = WiouB + (size_t)(H_ + n0) * X_ + kb16;
  const bf16_t* pWu = WiouB + (size_t)(2 * H_ + n0) * X_ + kb16;
  const bf16_t* pWf = WfB + (size_t)(n0)*X_ + kb16;
  const bf16_t* pA = hL + (lane & 15) * H_ + (lane >> 4) * 8;   // A from LDS
  const bf16_t* featRow =
      featB + (size_t)(b0 + (lane & 15)) * L_ * X_ + (lane >> 4) * 8;

  const v8f zacc = {0.f, 0.f, 0.f, 0.f, 0.f, 0.f, 0.f, 0.f};
  v8f cc = zacc;                // cell state for this wave's 16 columns
  v8f svI = zacc, svO = zacc, svU = zacc;  // iou_n saved at sweep1 l==0

  // acc: 0=i 1=o 2=u 3=f
  auto full_step = [&](int l, bool save) {
    v8f aI = zacc, aO = zacc, aU = zacc, aF = zacc;
    // recurrent part: h_p from LDS, U weights from L2
#pragma unroll
    for (int kk = 0; kk < 8; ++kk) {
      v16bf a = load_a_at(pA, kk);
      aI = wmma_bf16(a, load_b_at(pUi, kk), aI);
      aO = wmma_bf16(a, load_b_at(pUo, kk), aO);
      aU = wmma_bf16(a, load_b_at(pUu, kk), aU);
      aF = wmma_bf16(a, load_b_at(pUf, kk), aF);
    }
    if (save) { svI = aI; svO = aO; svU = aU; }
    // fused input projection
    const bf16_t* fr = featRow + l * X_;
    __builtin_prefetch(fr + X_, 0, 0);
#pragma unroll
    for (int kk = 0; kk < 4; ++kk) {
      v16bf a = load_a_at(fr, kk);
      aI = wmma_bf16(a, load_b_at(pWi, kk), aI);
      aO = wmma_bf16(a, load_b_at(pWo, kk), aO);
      aU = wmma_bf16(a, load_b_at(pWu, kk), aU);
      aF = wmma_bf16(a, load_b_at(pWf, kk), aF);
    }
    __syncthreads();  // all hL reads complete before overwrite
#pragma unroll
    for (int r = 0; r < 8; ++r) {
      float fg = sigm(aF[r] + bf);
      float c  = sigm(aI[r] + bi) * tanhf(aU[r] + bu) + fg * cc[r];
      cc[r] = c;
      hL[(r + rowOff) * H_ + n0] = (bf16_t)(sigm(aO[r] + bo) * tanhf(c));
    }
    __syncthreads();
  };

  auto hsc_step = [&](int l) {
    int rr = tid >> 4, seg = tid & 15;
    float p = 0.f;
    if (tid < 256) {
#pragma unroll
      for (int j = 0; j < 16; ++j)
        p += (float)hL[rr * H_ + seg * 16 + j] * lwL[seg * 16 + j];
      red[rr][seg] = p;
    }
    __syncthreads();
    if (tid < 256 && seg == 0) {
      float s = linb;
#pragma unroll
      for (int j = 0; j < 16; ++j) s += red[rr][j];
      hsc[(size_t)(b0 + rr) * L_ + l] = s;
    }
    __syncthreads();
  };

  // ---- sweep 1 (reversed), first step: original index L-1 ----
  {
    v8f aI, aO, aU;
#pragma unroll
    for (int r = 0; r < 8; ++r) {
      size_t rowbase = ((size_t)(b0 + r + rowOff) * L_ + (L_ - 1)) * H3_;
      aI[r] = iou_in[rowbase + n0];
      aO[r] = iou_in[rowbase + H_ + n0];
      aU[r] = iou_in[rowbase + 2 * H_ + n0];
    }
    const bf16_t* fr = featRow + (L_ - 1) * X_;
#pragma unroll
    for (int kk = 0; kk < 4; ++kk) {
      v16bf a = load_a_at(fr, kk);
      aI = wmma_bf16(a, load_b_at(pWi, kk), aI);
      aO = wmma_bf16(a, load_b_at(pWo, kk), aO);
      aU = wmma_bf16(a, load_b_at(pWu, kk), aU);
    }
#pragma unroll
    for (int r = 0; r < 8; ++r) {
      int m = r + rowOff;
      float cr = c_in[((size_t)(b0 + m) * L_ + (L_ - 1)) * H_ + n0];
      float c  = sigm(aI[r] + bi) * tanhf(aU[r] + bu) + cr;
      cc[r] = c;
      hL[m * H_ + n0] = (bf16_t)(sigm(aO[r] + bo) * tanhf(c));
    }
    __syncthreads();
  }

  // ---- sweep 1 remaining steps: original indices L-2 .. 0 ----
  for (int l = L_ - 2; l >= 0; --l) full_step(l, l == 0);

  // ---- sweep 2, l = 0: tot = Wx[:,0] + b_iou + saved iou_n; c_red = c ----
  {
    v8f aI = svI, aO = svO, aU = svU;
#pragma unroll
    for (int kk = 0; kk < 4; ++kk) {
      v16bf a = load_a_at(featRow, kk);   // l = 0
      aI = wmma_bf16(a, load_b_at(pWi, kk), aI);
      aO = wmma_bf16(a, load_b_at(pWo, kk), aO);
      aU = wmma_bf16(a, load_b_at(pWu, kk), aU);
    }
    __syncthreads();
#pragma unroll
    for (int r = 0; r < 8; ++r) {
      float c = sigm(aI[r] + bi) * tanhf(aU[r] + bu) + cc[r];
      cc[r] = c;
      hL[(r + rowOff) * H_ + n0] = (bf16_t)(sigm(aO[r] + bo) * tanhf(c));
    }
    __syncthreads();
    hsc_step(0);
  }

  // ---- sweep 2 remaining steps with fused linear score ----
  for (int l = 1; l < L_; ++l) {
    full_step(l, false);
    hsc_step(l);
  }
}

// ---------------------------------------------------------------------------
__global__ void score_accum(const float* __restrict__ hsc,
                            const float* __restrict__ sid,
                            const float* __restrict__ siu,
                            const int* __restrict__ var_idx,
                            float* __restrict__ bins) {
  __shared__ float sc[64], sd[64], su[64];
  int t = threadIdx.x;
  if (t < 64) { sc[t] = 0.f; sd[t] = 0.f; su[t] = 0.f; }
  __syncthreads();
  for (int i = blockIdx.x * blockDim.x + t; i < B_ * L_;
       i += gridDim.x * blockDim.x) {
    int v = var_idx[i];
    if (v >= 0 && v < 64) {
      float hv = hsc[i];
      atomicAdd(&sc[v], 1.0f);
      atomicAdd(&sd[v], hv * sid[i]);
      atomicAdd(&su[v], hv * siu[i]);
    }
  }
  __syncthreads();
  if (t < 64) {
    atomicAdd(&bins[t], sc[t]);
    atomicAdd(&bins[64 + t], sd[t]);
    atomicAdd(&bins[128 + t], su[t]);
  }
}

__global__ void finalize(const float* __restrict__ bins,
                         const int* __restrict__ cand,
                         float* __restrict__ out) {
  __shared__ float sc[K_];
  int t = threadIdx.x;
  if (t < K_) {
    int cv = cand[t];
    float cnt = bins[cv];
    float den = fmaxf(cnt, 1.0f);
    float pd = bins[64 + cv] / den;
    float pu = bins[128 + cv] / den;
    float s = (cnt > 0.f) ? (0.5f * pd + 0.5f * fmaxf(pd, pu)) : 0.f;
    sc[t] = s;
    out[1 + t] = s;
  }
  __syncthreads();
  if (t == 0) {
    int best = 0;
    float bv = sc[0];
    for (int k = 1; k < K_; ++k)
      if (sc[k] > bv) { bv = sc[k]; best = k; }
    out[0] = (float)best;
  }
}

// ---------------------------------------------------------------------------
extern "C" void kernel_launch(void* const* d_in, const int* in_sizes, int n_in,
                              void* d_out, int out_size, void* d_ws,
                              size_t ws_size, hipStream_t stream) {
  const float* features = (const float*)d_in[0];
  // d_in[1] (h) is unused by the reference
  const float* c_in    = (const float*)d_in[2];
  const float* iou_in  = (const float*)d_in[3];
  const float* sid     = (const float*)d_in[4];
  const float* siu     = (const float*)d_in[5];
  const int*   var_idx = (const int*)d_in[6];
  const int*   cand    = (const int*)d_in[7];
  const float* W_iou   = (const float*)d_in[8];
  const float* U_iou   = (const float*)d_in[9];
  const float* b_iou   = (const float*)d_in[10];
  const float* W_f     = (const float*)d_in[11];
  const float* bW_f    = (const float*)d_in[12];
  const float* b_f     = (const float*)d_in[13];
  const float* U_f     = (const float*)d_in[14];
  const float* bU_f    = (const float*)d_in[15];
  const float* lin_w   = (const float*)d_in[16];
  const float* lin_b   = (const float*)d_in[17];

  char* ws   = (char*)d_ws;
  size_t off = 0;
  auto carve = [&](size_t bytes) -> char* {
    char* p = ws + off;
    off += (bytes + 255) & ~(size_t)255;
    return p;
  };
  bf16_t* featB = (bf16_t*)carve((size_t)B_ * L_ * X_ * 2);
  bf16_t* WiouB = (bf16_t*)carve((size_t)H3_ * X_ * 2);
  bf16_t* UiouB = (bf16_t*)carve((size_t)H3_ * H_ * 2);
  bf16_t* WfB   = (bf16_t*)carve((size_t)H_ * X_ * 2);
  bf16_t* UfB   = (bf16_t*)carve((size_t)H_ * H_ * 2);
  float*  hsc   = (float*)carve((size_t)B_ * L_ * 4);
  float*  bins  = (float*)carve(192 * 4);

  // f32 -> bf16 staging (features + weights)
  cvt_f32_bf16<<<1024, 256, 0, stream>>>(features, featB, B_ * L_ * X_);
  cvt_f32_bf16<<<96, 256, 0, stream>>>(W_iou, WiouB, H3_ * X_);
  cvt_f32_bf16<<<192, 256, 0, stream>>>(U_iou, UiouB, H3_ * H_);
  cvt_f32_bf16<<<32, 256, 0, stream>>>(W_f, WfB, H_ * X_);
  cvt_f32_bf16<<<64, 256, 0, stream>>>(U_f, UfB, H_ * H_);

  zero_bins<<<1, 192, 0, stream>>>(bins);

  treelstm_sweep<<<B_ / 16, 512, 0, stream>>>(
      featB, WiouB, UiouB, WfB, UfB, iou_in, c_in, b_iou, bW_f, b_f, bU_f,
      lin_w, lin_b, hsc);

  score_accum<<<128, 256, 0, stream>>>(hsc, sid, siu, var_idx, bins);
  finalize<<<1, 64, 0, stream>>>(bins, cand, (float*)d_out);
}
